// LSTMClassifier_7224134992379
// MI455X (gfx1250) — matrix operand order repaired and checked by reading, not compiled, not run on hardware
//
#include <hip/hip_runtime.h>

#define T_DIM 1024
#define B_DIM 128
#define I_DIM 256
#define H_DIM 512
#define G_DIM 2048
#define H_PAD (H_DIM + 8)   // LDS bank-conflict padding (row stride 1040B, 16B-aligned)

typedef __bf16 bf16_t;
typedef __attribute__((ext_vector_type(16))) __bf16         v16bf;
typedef __attribute__((ext_vector_type(8)))  float          v8f;
typedef __attribute__((ext_vector_type(8)))  unsigned short v8u16;
typedef __attribute__((ext_vector_type(16))) unsigned short v16u16;

__device__ __forceinline__ bf16_t to_bf16(float f) {
  unsigned int u = __float_as_uint(f);
  unsigned int rnd = 0x7FFFu + ((u >> 16) & 1u);   // round-to-nearest-even
  unsigned short h = (unsigned short)((u + rnd) >> 16);
  return __builtin_bit_cast(bf16_t, h);
}

__device__ __forceinline__ v16bf make_frag(v8u16 lo, v8u16 hi) {
  v16u16 t;
#pragma unroll
  for (int e = 0; e < 8; ++e) { t[e] = lo[e]; t[e + 8] = hi[e]; }
  return __builtin_bit_cast(v16bf, t);
}

__device__ __forceinline__ float sigmoidf_(float x) {
  return 1.0f / (1.0f + __expf(-x));
}

// ---------------------------------------------------------------------------
// f32 -> bf16 conversion (x and the four weight matrices)
// ---------------------------------------------------------------------------
__global__ void cvt_f32_bf16(const float* __restrict__ in, bf16_t* __restrict__ out, int n) {
  int i = blockIdx.x * blockDim.x + threadIdx.x;
  if (i < n) out[i] = to_bf16(in[i]);
}

// ---------------------------------------------------------------------------
// Gate pre-activation GEMM:  out[t,b,:] = A[b*T+t,:] @ W^T + (b_ih + b_hh)
// A: [M=B*T, K] bf16 row-major;  W: [G, K] bf16 row-major;  out: [T, B, G] f32
// One wave computes a 16(M) x 64(N) slab: A-frag reused across 4 WMMA N-tiles.
// ---------------------------------------------------------------------------
template <int K>
__global__ __launch_bounds__(256) void gemm_gates(
    const bf16_t* __restrict__ A, const bf16_t* __restrict__ W,
    const float* __restrict__ bih, const float* __restrict__ bhh,
    float* __restrict__ out) {
  const int lane = threadIdx.x & 31;
  const int wv   = threadIdx.x >> 5;       // 8 waves / block
  const int half = lane >> 4;              // 0 or 1 (lane group)
  const int r    = lane & 15;

  const int tiles_n = G_DIM / 64;          // 32
  const int gt = blockIdx.x * 8 + wv;
  const size_t m0 = (size_t)(gt / tiles_n) * 16;
  const int    n0 = (gt % tiles_n) * 64;

  v8f zero = {};
  v8f acc[4] = {zero, zero, zero, zero};

  const bf16_t* arow = A + (m0 + r) * K;   // A row for this lane (16B-aligned chunks)

  for (int kk = 0; kk < K / 32; ++kk) {
    // A 16x32 bf16 fragment: per-lane 16 contiguous K at half*16
    const v8u16* pa = (const v8u16*)(arow + kk * 32 + half * 16);
    v16bf afrag = make_frag(pa[0], pa[1]);
#pragma unroll
    for (int j = 0; j < 4; ++j) {
      // B 32x16 bf16 fragment: col = n+r, 16 contiguous K at half*16
      const v8u16* pb =
          (const v8u16*)(W + (size_t)(n0 + j * 16 + r) * K + kk * 32 + half * 16);
      v16bf bfrag = make_frag(pb[0], pb[1]);
      acc[j] = __builtin_amdgcn_wmma_f32_16x16x32_bf16(
          false, afrag, false, bfrag, (short)0, acc[j], false, false);
    }
  }

#pragma unroll
  for (int j = 0; j < 4; ++j) {
    const int col = n0 + j * 16 + r;
    const float bias = bih[col] + bhh[col];
#pragma unroll
    for (int e = 0; e < 8; ++e) {
      const size_t m = m0 + e + half * 8;          // global row = b*T + t
      const size_t b = m / T_DIM, t = m % T_DIM;
      out[(t * B_DIM + b) * G_DIM + col] = acc[j][e] + bias;
    }
  }
}

// ---------------------------------------------------------------------------
// Sequential LSTM recurrence for one layer.
// Grid: 8 workgroups (one per 16-row batch slab) x 1024 threads (32 waves).
// Wave w owns the 16x16 hidden tile at columns [w*16, w*16+16).
// h (16x512 bf16) lives in LDS, shared across all 32 waves; c stays in VGPRs.
// W_hh bf16 (2MB) is L2-resident; streamed every step.
// ---------------------------------------------------------------------------
template <bool WRITE_SEQ, bool WRITE_LAST>
__global__ __launch_bounds__(1024) void lstm_recurrence(
    const float* __restrict__ xg,       // [T, B, G] gate pre-activations
    const bf16_t* __restrict__ Whh,     // [G, H] bf16
    bf16_t* __restrict__ out_seq,       // [B, T, H] bf16 (layer-0 output)
    float* __restrict__ h_last) {       // [B, H] f32   (layer-1 final h)
  __shared__ bf16_t h_lds[16 * H_PAD];  // ~16.6 KB of 320 KB/WGP

  const int lane = threadIdx.x & 31;
  const int wv   = threadIdx.x >> 5;    // 0..31 -> hidden column tile
  const int half = lane >> 4;
  const int r    = lane & 15;
  const int b0   = blockIdx.x * 16;     // batch-row base of this workgroup
  const int n_h  = wv * 16;             // hidden-column base of this wave

  // h_0 = 0
  for (int i = threadIdx.x; i < 16 * H_PAD; i += 1024) h_lds[i] = to_bf16(0.0f);
  __syncthreads();

  v8f c_acc = {};                       // cell state, f32, in registers

#pragma unroll 1
  for (int t = 0; t < T_DIM; ++t) {
    const float* xg_t = xg + (size_t)t * B_DIM * G_DIM;

    // Seed accumulators with x-contribution (gate order: i, f, g, o)
    v8f acc[4];
#pragma unroll
    for (int g = 0; g < 4; ++g) {
      const int col = g * H_DIM + n_h + r;
      v8f a;
#pragma unroll
      for (int e = 0; e < 8; ++e)
        a[e] = xg_t[(size_t)(b0 + e + half * 8) * G_DIM + col];
      acc[g] = a;
    }

    // gates += h_prev @ W_hh^T   (K = 512, A-frag from LDS reused over 4 gates)
#pragma unroll 4
    for (int kk = 0; kk < H_DIM / 32; ++kk) {
      const v8u16* pa = (const v8u16*)(&h_lds[r * H_PAD + kk * 32 + half * 16]);
      v16bf afrag = make_frag(pa[0], pa[1]);
#pragma unroll
      for (int g = 0; g < 4; ++g) {
        const v8u16* pb = (const v8u16*)(Whh + (size_t)(g * H_DIM + n_h + r) * H_DIM +
                                         kk * 32 + half * 16);
        v16bf bfrag = make_frag(pb[0], pb[1]);
        acc[g] = __builtin_amdgcn_wmma_f32_16x16x32_bf16(
            false, afrag, false, bfrag, (short)0, acc[g], false, false);
      }
    }

    __syncthreads();  // all waves done reading h_{t-1} from LDS

    // Pointwise gate math; write h_t
    v8f h_new;
#pragma unroll
    for (int e = 0; e < 8; ++e) {
      const float ig = sigmoidf_(acc[0][e]);
      const float fg = sigmoidf_(acc[1][e]);
      const float gg = tanhf(acc[2][e]);
      const float og = sigmoidf_(acc[3][e]);
      const float c  = fg * c_acc[e] + ig * gg;
      c_acc[e] = c;
      h_new[e] = og * tanhf(c);
    }
#pragma unroll
    for (int e = 0; e < 8; ++e)
      h_lds[(e + half * 8) * H_PAD + n_h + r] = to_bf16(h_new[e]);

    if (WRITE_SEQ) {
#pragma unroll
      for (int e = 0; e < 8; ++e) {
        const size_t b = b0 + e + half * 8;
        out_seq[(b * T_DIM + t) * H_DIM + n_h + r] = to_bf16(h_new[e]);
      }
    }
    if (WRITE_LAST && t == T_DIM - 1) {
#pragma unroll
      for (int e = 0; e < 8; ++e)
        h_last[(size_t)(b0 + e + half * 8) * H_DIM + n_h + r] = h_new[e];
    }

    __syncthreads();  // h_t visible before next step's reads
  }
}

// ---------------------------------------------------------------------------
// Final FC: out[b] = h_last[b,:] . fc_w + fc_b   (tiny: 128x512)
// ---------------------------------------------------------------------------
__global__ __launch_bounds__(64) void fc_kernel(const float* __restrict__ h,
                                                const float* __restrict__ w,
                                                const float* __restrict__ bias,
                                                float* __restrict__ out) {
  __shared__ float red[64];
  const int b = blockIdx.x;
  float s = 0.0f;
  for (int i = threadIdx.x; i < H_DIM; i += 64) s += h[(size_t)b * H_DIM + i] * w[i];
  red[threadIdx.x] = s;
  __syncthreads();
  for (int off = 32; off > 0; off >>= 1) {
    if (threadIdx.x < off) red[threadIdx.x] += red[threadIdx.x + off];
    __syncthreads();
  }
  if (threadIdx.x == 0) out[b] = red[0] + bias[0];
}

// ---------------------------------------------------------------------------
extern "C" void kernel_launch(void* const* d_in, const int* in_sizes, int n_in,
                              void* d_out, int out_size, void* d_ws, size_t ws_size,
                              hipStream_t stream) {
  const float* x    = (const float*)d_in[0];
  const float* Wih0 = (const float*)d_in[1];
  const float* Whh0 = (const float*)d_in[2];
  const float* bih0 = (const float*)d_in[3];
  const float* bhh0 = (const float*)d_in[4];
  const float* Wih1 = (const float*)d_in[5];
  const float* Whh1 = (const float*)d_in[6];
  const float* bih1 = (const float*)d_in[7];
  const float* bhh1 = (const float*)d_in[8];
  const float* fcw  = (const float*)d_in[9];
  const float* fcb  = (const float*)d_in[10];
  float* out = (float*)d_out;

  // Carve workspace (256B aligned sub-allocations)
  char* p = (char*)d_ws;
  auto carve = [&](size_t bytes) {
    char* q = p;
    p += (bytes + 255) & ~(size_t)255;
    return (void*)q;
  };
  float*  xg    = (float*)carve((size_t)T_DIM * B_DIM * G_DIM * 4);   // 1 GB, reused per layer
  bf16_t* xb    = (bf16_t*)carve((size_t)B_DIM * T_DIM * I_DIM * 2);  // 64 MB
  bf16_t* o0    = (bf16_t*)carve((size_t)B_DIM * T_DIM * H_DIM * 2);  // 128 MB
  bf16_t* wih0b = (bf16_t*)carve((size_t)G_DIM * I_DIM * 2);
  bf16_t* whh0b = (bf16_t*)carve((size_t)G_DIM * H_DIM * 2);
  bf16_t* wih1b = (bf16_t*)carve((size_t)G_DIM * H_DIM * 2);
  bf16_t* whh1b = (bf16_t*)carve((size_t)G_DIM * H_DIM * 2);
  float*  hlast = (float*)carve((size_t)B_DIM * H_DIM * 4);

  auto cvt = [&](const float* src, bf16_t* dst, int n) {
    cvt_f32_bf16<<<(n + 255) / 256, 256, 0, stream>>>(src, dst, n);
  };
  cvt(x,    xb,    B_DIM * T_DIM * I_DIM);
  cvt(Wih0, wih0b, G_DIM * I_DIM);
  cvt(Whh0, whh0b, G_DIM * H_DIM);
  cvt(Wih1, wih1b, G_DIM * H_DIM);
  cvt(Whh1, whh1b, G_DIM * H_DIM);

  const int gemm_blocks = (B_DIM * T_DIM / 16) * (G_DIM / 64) / 8;  // 32768

  // Layer 0
  gemm_gates<I_DIM><<<gemm_blocks, 256, 0, stream>>>(xb, wih0b, bih0, bhh0, xg);
  lstm_recurrence<true, false><<<B_DIM / 16, 1024, 0, stream>>>(xg, whh0b, o0, nullptr);

  // Layer 1
  gemm_gates<H_DIM><<<gemm_blocks, 256, 0, stream>>>(o0, wih1b, bih1, bhh1, xg);
  lstm_recurrence<false, true><<<B_DIM / 16, 1024, 0, stream>>>(xg, whh1b, nullptr, hlast);

  // Classifier head
  fc_kernel<<<B_DIM, 64, 0, stream>>>(hlast, fcw, fcb, out);
}
